// WindowedAttention_73856257622522
// MI455X (gfx1250) — compile-verified, matrix operand order checked by reading, not run on hardware
//
#include <hip/hip_runtime.h>
#include <hip/hip_bf16.h>

typedef __attribute__((ext_vector_type(16))) __bf16 v16bf;
typedef __attribute__((ext_vector_type(8)))  __bf16 v8bf;
typedef __attribute__((ext_vector_type(8)))  float  v8f;

#define WS_       8
#define NTOK      64          // tokens per window
#define DIM_      128
#define NH_       4
#define HD_       32
#define SCALE_    0.17677669529663689f   // 1/sqrt(32)
#define XPITCH    136         // bf16 pitch for 128-wide rows
#define QKPITCH   40          // bf16 pitch for 32-wide rows
#define APITCH    72          // bf16 pitch for 64-wide rows

// ---------------- helpers ----------------

// Load a 16x32 bf16 WMMA fragment's per-lane 16 halfs: two contiguous
// 8-half (16B) chunks at [kb, kb+8) and [kb+16, kb+24).
__device__ __forceinline__ v16bf frag16(const __bf16* p) {
    v8bf lo = *reinterpret_cast<const v8bf*>(p);
    v8bf hi = *reinterpret_cast<const v8bf*>(p + 16);
    v16bf a;
#pragma unroll
    for (int i = 0; i < 8; ++i) { a[i] = lo[i]; a[8 + i] = hi[i]; }
    return a;
}

__device__ __forceinline__ v8f wmma_bf16(v16bf a, v16bf b, v8f c) {
    return __builtin_amdgcn_wmma_f32_16x16x32_bf16(
        /*neg_a=*/false, a, /*neg_b=*/false, b,
        /*c_mod=*/(short)0, c, /*reuse_a=*/false, /*reuse_b=*/false);
}

// ---------------- weight prep: fp32 -> bf16 ----------------
__global__ void WindowedAttention_prep_kernel(const float* __restrict__ qkvw,
                                              const float* __restrict__ projw,
                                              __bf16* __restrict__ Wq,
                                              __bf16* __restrict__ Wp) {
    int i = blockIdx.x * 256 + threadIdx.x;      // 65536 total
    if (i < 3 * DIM_ * DIM_) {
        Wq[i] = (__bf16)qkvw[i];
    } else {
        int j = i - 3 * DIM_ * DIM_;
        Wp[j] = (__bf16)projw[j];
    }
}

// ---------------- fused per-window attention ----------------
__global__ __launch_bounds__(256)
void WindowedAttention_73856257622522_kernel(const float* __restrict__ x,
                                             const float* __restrict__ qkvb,
                                             const float* __restrict__ projb,
                                             const float* __restrict__ rpb,
                                             const __bf16* __restrict__ Wq,
                                             const __bf16* __restrict__ Wp,
                                             float* __restrict__ out) {
    __shared__ alignas(16) __bf16 Xs[NTOK * XPITCH];            // x window, bf16
    __shared__ alignas(16) __bf16 qs[NH_ * NTOK * QKPITCH];     // q * scale
    __shared__ alignas(16) __bf16 ks[NH_ * NTOK * QKPITCH];     // k
    __shared__ alignas(16) __bf16 vT[NH_ * HD_ * APITCH];       // v transposed [d][tok]
    __shared__ alignas(16) __bf16 attnP[NH_ * NTOK * APITCH];   // softmax probs
    __shared__ alignas(16) __bf16 outA[NTOK * XPITCH];          // attn output [tok][C]

    const int tid  = threadIdx.x;
    const int lane = tid & 31;
    const int wv   = tid >> 5;          // 8 waves
    const int w    = blockIdx.x;        // window id, 8192 total
    const int b    = w >> 10;
    const int rem  = w & 1023;
    const int wr   = rem >> 5;
    const int wc   = rem & 31;

    const int lm   = lane & 15;         // n/m within tile
    const int kb0  = (lane >> 4) * 8;   // per-lane k chunk base
    const int rhal = (lane >> 4) << 3;  // +8 row offset for upper half lanes

    // warm L1/L2 for the bf16 weights (global_prefetch_b8)
    __builtin_prefetch(Wq + (size_t)tid * 192, 0, 3);
    __builtin_prefetch(Wp + (size_t)tid * 64, 0, 3);

    // ---- Phase 0: stage x window into LDS as bf16 ----
    {
        int t  = tid >> 2;                  // token 0..63
        int c0 = (tid & 3) * 32;            // 32 channels per thread
        int grow = wr * WS_ + (t >> 3);
        int gcol = wc * WS_ + (t & 7);
        const float4* xp = reinterpret_cast<const float4*>(
            x + ((((size_t)b * 256 + grow) * 256) + gcol) * DIM_ + c0);
#pragma unroll
        for (int i = 0; i < 8; ++i) {
            float4 f = xp[i];
            __bf16* d = &Xs[t * XPITCH + c0 + i * 4];
            d[0] = (__bf16)f.x; d[1] = (__bf16)f.y;
            d[2] = (__bf16)f.z; d[3] = (__bf16)f.w;
        }
    }
    __syncthreads();

    // ---- Phase 1: QKV GEMM  [64x128] @ [128x384]^T ----
    // wave -> fixed m-tile (mt = wv/2), 12 n-tiles; A-fragments hoisted.
    {
        int mt   = wv >> 1;
        int ntb  = (wv & 1) * 12;
        int arow = mt * 16 + lm;
        v16bf aF[4];
#pragma unroll
        for (int kt = 0; kt < 4; ++kt)
            aF[kt] = frag16(&Xs[arow * XPITCH + kt * 32 + kb0]);

#pragma unroll
        for (int i = 0; i < 12; ++i) {
            int nt = ntb + i;
            int n  = nt * 16 + lm;          // output column 0..383
            float bn = qkvb[n];
            v8f acc;
#pragma unroll
            for (int r = 0; r < 8; ++r) acc[r] = bn;
#pragma unroll
            for (int kt = 0; kt < 4; ++kt) {
                v16bf bf = frag16(&Wq[n * DIM_ + kt * 32 + kb0]);
                acc = wmma_bf16(aF[kt], bf, acc);
            }
            int sect = n >> 7;              // 0=q 1=k 2=v (uniform per tile)
            int h    = (n & 127) >> 5;
            int d    = n & 31;
            int rb   = mt * 16 + rhal;
            if (sect == 0) {
#pragma unroll
                for (int r = 0; r < 8; ++r)
                    qs[(h * NTOK + rb + r) * QKPITCH + d] = (__bf16)(acc[r] * SCALE_);
            } else if (sect == 1) {
#pragma unroll
                for (int r = 0; r < 8; ++r)
                    ks[(h * NTOK + rb + r) * QKPITCH + d] = (__bf16)acc[r];
            } else {
#pragma unroll
                for (int r = 0; r < 8; ++r)
                    vT[(h * HD_ + d) * APITCH + rb + r] = (__bf16)acc[r];
            }
        }
    }
    __syncthreads();

    // ---- Phase 2: scores + rel-pos bias + softmax (16 row-blocks, 2/wave) ----
#pragma unroll
    for (int j = 0; j < 2; ++j) {
        int rb = wv * 2 + j;
        int h  = rb >> 2, mt = rb & 3;
        int mrow = mt * 16 + lm;
        v16bf aq = frag16(&qs[(h * NTOK + mrow) * QKPITCH + kb0]);  // K=32: 1 step
        int ti = mt * 16 + rhal;            // C-layout row base
        v8f sc[4];
#pragma unroll
        for (int nt = 0; nt < 4; ++nt) {
            int ncol = nt * 16 + lm;
            v8f c;
#pragma unroll
            for (int r = 0; r < 8; ++r) {   // rel-pos bias as C operand
                int irow = ti + r;
                int idx = ((irow >> 3) - (ncol >> 3) + 7) * 15
                        + ((irow & 7) - (ncol & 7) + 7);
                c[r] = rpb[idx * NH_ + h];
            }
            v16bf bk = frag16(&ks[(h * NTOK + ncol) * QKPITCH + kb0]);
            sc[nt] = wmma_bf16(aq, bk, c);
        }
        // softmax: each row lives in 16 lanes (one half-wave), 4 cols/lane
#pragma unroll
        for (int r = 0; r < 8; ++r) {
            float s0 = sc[0][r], s1 = sc[1][r], s2 = sc[2][r], s3 = sc[3][r];
            float mx = fmaxf(fmaxf(s0, s1), fmaxf(s2, s3));
            mx = fmaxf(mx, __shfl_xor(mx, 1, 32));
            mx = fmaxf(mx, __shfl_xor(mx, 2, 32));
            mx = fmaxf(mx, __shfl_xor(mx, 4, 32));
            mx = fmaxf(mx, __shfl_xor(mx, 8, 32));
            float p0 = __expf(s0 - mx), p1 = __expf(s1 - mx);
            float p2 = __expf(s2 - mx), p3 = __expf(s3 - mx);
            float sm = p0 + p1 + p2 + p3;
            sm += __shfl_xor(sm, 1, 32);
            sm += __shfl_xor(sm, 2, 32);
            sm += __shfl_xor(sm, 4, 32);
            sm += __shfl_xor(sm, 8, 32);
            float inv = 1.0f / sm;
            int row = ti + r;
            __bf16* dst = &attnP[(h * NTOK + row) * APITCH + lm];
            dst[0]  = (__bf16)(p0 * inv);
            dst[16] = (__bf16)(p1 * inv);
            dst[32] = (__bf16)(p2 * inv);
            dst[48] = (__bf16)(p3 * inv);
        }
    }
    __syncthreads();

    // ---- Phase 3: attn @ V  (wave -> head h=wv/2, two m-tiles; B hoisted) ----
    {
        int h   = wv >> 1;
        int sub = wv & 1;                   // m-tile pair selector
        v16bf bF[2][2];                     // [nt][kt], reused across m-tiles
#pragma unroll
        for (int nt = 0; nt < 2; ++nt)
#pragma unroll
            for (int kt = 0; kt < 2; ++kt)
                bF[nt][kt] = frag16(&vT[(h * HD_ + nt * 16 + lm) * APITCH
                                        + kt * 32 + kb0]);
#pragma unroll
        for (int mi = 0; mi < 2; ++mi) {
            int mt   = sub * 2 + mi;
            int mrow = mt * 16 + lm;
            v16bf a0 = frag16(&attnP[(h * NTOK + mrow) * APITCH + kb0]);
            v16bf a1 = frag16(&attnP[(h * NTOK + mrow) * APITCH + 32 + kb0]);
            int rb = mt * 16 + rhal;
#pragma unroll
            for (int nt = 0; nt < 2; ++nt) {
                v8f acc;
#pragma unroll
                for (int r = 0; r < 8; ++r) acc[r] = 0.0f;
                acc = wmma_bf16(a0, bF[nt][0], acc);
                acc = wmma_bf16(a1, bF[nt][1], acc);
#pragma unroll
                for (int r = 0; r < 8; ++r)
                    outA[(rb + r) * XPITCH + h * HD_ + nt * 16 + lm] = (__bf16)acc[r];
            }
        }
    }
    __syncthreads();

    // ---- Phase 4: output projection  [64x128] @ [128x128]^T ----
    // wave -> fixed m-tile (mt = wv/2), 4 n-tiles; A-fragments hoisted.
    {
        int mt   = wv >> 1;
        int ntb  = (wv & 1) * 4;
        int mrow = mt * 16 + lm;
        v16bf aF[4];
#pragma unroll
        for (int kt = 0; kt < 4; ++kt)
            aF[kt] = frag16(&outA[mrow * XPITCH + kt * 32 + kb0]);

        int rb = mt * 16 + rhal;
        size_t obase = (size_t)w * (NTOK * DIM_);
#pragma unroll
        for (int i = 0; i < 4; ++i) {
            int nt = ntb + i;
            int n  = nt * 16 + lm;
            float bn = projb[n];
            v8f acc;
#pragma unroll
            for (int r = 0; r < 8; ++r) acc[r] = bn;
#pragma unroll
            for (int kt = 0; kt < 4; ++kt) {
                v16bf bf = frag16(&Wp[n * DIM_ + kt * 32 + kb0]);
                acc = wmma_bf16(aF[kt], bf, acc);
            }
#pragma unroll
            for (int r = 0; r < 8; ++r)
                out[obase + (size_t)(rb + r) * DIM_ + n] = acc[r];   // coalesced
        }
    }
}

extern "C" void kernel_launch(void* const* d_in, const int* in_sizes, int n_in,
                              void* d_out, int out_size, void* d_ws, size_t ws_size,
                              hipStream_t stream) {
    const float* x     = (const float*)d_in[0];
    const float* qkvw  = (const float*)d_in[1];
    const float* qkvb  = (const float*)d_in[2];
    const float* projw = (const float*)d_in[3];
    const float* projb = (const float*)d_in[4];
    const float* rpb   = (const float*)d_in[5];
    float* out = (float*)d_out;

    __bf16* Wq = (__bf16*)d_ws;                    // 3*128*128 bf16 = 96 KB
    __bf16* Wp = Wq + 3 * DIM_ * DIM_;             // 128*128 bf16   = 32 KB

    // 65536 weight elements, 1 per thread
    WindowedAttention_prep_kernel<<<256, 256, 0, stream>>>(qkvw, projw, Wq, Wp);

    // one workgroup (8 wave32) per 8x8 window; 8 * 32 * 32 = 8192 windows
    WindowedAttention_73856257622522_kernel<<<8192, 256, 0, stream>>>(
        x, qkvb, projb, rpb, Wq, Wp, out);
}
